// Model_22488448762490
// MI455X (gfx1250) — matrix operand order repaired and checked by reading, not compiled, not run on hardware
//
#include <hip/hip_runtime.h>
#include <hip/hip_bf16.h>
#include <math.h>

// ---------------- problem constants (from reference setup) ----------------
#define HID   51      // real hidden size
#define BATCH 1024
#define TLEN  1024
#define BT    16      // batch rows per workgroup (one WMMA M-tile)
#define KP    64      // padded hidden size
#define NP    256     // padded gate dim = 4 gates * 64
#define NTH   128     // 4 waves; wave w owns j-slice w (16 cols) across all 4 gates

typedef __attribute__((ext_vector_type(16))) _Float16 v16h;
typedef __attribute__((ext_vector_type(8)))  _Float16 v8h;
typedef __attribute__((ext_vector_type(8)))  float    v8f;

union AU { v16h v; v8h p[2]; _Float16 h[16]; };
union CU { v8f  v; float f[8]; };

// Branch-free fast activations: v_exp_f32 + v_rcp_f32, no exec divergence.
__device__ __forceinline__ float sigf(float x) {
  return __builtin_amdgcn_rcpf(1.0f + __expf(-x));
}
__device__ __forceinline__ float tanh_fast(float x) {
  // tanh(x) = 1 - 2/(exp(2x)+1); saturates correctly for |x| large.
  return 1.0f - 2.0f * __builtin_amdgcn_rcpf(1.0f + __expf(2.0f * x));
}

// Padded weight accessors: gates[m][n] = sum_k h[m][k]*W[n][k] => B[k][n] = W[n][k]
// n = g*64 + j; zero outside real dims so padding stays exactly zero.
__device__ __forceinline__ float w1_at(const float* __restrict__ Whh1, int k, int n) {
  int g = n >> 6, j = n & 63;
  if (j >= HID || k >= HID) return 0.0f;
  return Whh1[(g * HID + j) * HID + k];
}
__device__ __forceinline__ float w2_at(const float* __restrict__ Wih2,
                                       const float* __restrict__ Whh2, int k, int n) {
  int g = n >> 6, j = n & 63;
  if (j >= HID) return 0.0f;
  int r = g * HID + j;
  if (k < KP) return (k < HID) ? Wih2[r * HID + k] : 0.0f;
  k -= KP;    return (k < HID) ? Whh2[r * HID + k] : 0.0f;
}

__device__ __forceinline__ v8f wmma_f16(const AU& a, const AU& b, v8f c) {
  return __builtin_amdgcn_wmma_f32_16x16x32_f16(false, a.v, false, b.v,
                                                (short)0, c, false, false);
}

__global__ void __launch_bounds__(NTH)
lstm2_persistent(const float* __restrict__ x,
                 const float* __restrict__ W_ih1, const float* __restrict__ W_hh1,
                 const float* __restrict__ b_ih1, const float* __restrict__ b_hh1,
                 const float* __restrict__ W_ih2, const float* __restrict__ W_hh2,
                 const float* __restrict__ b_ih2, const float* __restrict__ b_hh2,
                 const float* __restrict__ W_lin, const float* __restrict__ b_lin,
                 const int* __restrict__ future_p,
                 float* __restrict__ out)
{
  // LDS: layer-1 weights (B-source) + ping-pong f16 hidden states.
  __shared__ __align__(16) _Float16 w1t[NP * KP];        // W1^T [n][k], 32 KB
  __shared__ __align__(16) _Float16 h1buf[2][BT * KP];   // 2 KB each
  __shared__ __align__(16) _Float16 h2buf[2][BT * KP];
  __shared__ __align__(16) float wlinpad[KP];
  __shared__ __align__(32) float xbuf[BT];
  __shared__ float blin;

  const int tid  = (int)threadIdx.x;
  const int lane = tid & 31;
  const int js   = tid >> 5;            // wave id == j-slice (0..3)
  const int b0   = (int)blockIdx.x * BT;
  const int future = *future_p;
  const int TF = TLEN + future;

  // ---------------- one-time init ----------------
  for (int e = tid; e < NP * KP; e += NTH) {
    int n = e >> 6, k = e & 63;
    w1t[e] = (_Float16)w1_at(W_hh1, k, n);
  }
  for (int e = tid; e < BT * KP; e += NTH) {
    h1buf[0][e] = (_Float16)0.0f; h1buf[1][e] = (_Float16)0.0f;
    h2buf[0][e] = (_Float16)0.0f; h2buf[1][e] = (_Float16)0.0f;
  }
  if (tid < KP) wlinpad[tid] = (tid < HID) ? W_lin[tid] : 0.0f;
  if (tid < BT) xbuf[tid] = x[(size_t)(b0 + tid) * TLEN + 0];
  if (tid == 0) blin = b_lin[0];

  // per-lane geometry
  const int nlan  = lane & 15;
  const int hi    = lane >> 4;          // 0/1 half of wave
  const int jcol  = js * 16 + nlan;     // column owned in C/activation layout
  const int klow  = hi * 8;             // B 32x16 layout: lanes 16-31 hold K 8..15, 24..31
  const int kl8   = hi;                 // klow in v8h units
  const int m_a   = nlan;               // A 16x32 layout: m = lane&15
  const int abase = hi;                 // kbase (0/8 halfs) in v8h units
  const int mrow0 = hi * 8;             // C layout: rows r -> M = r + 8*hi

  // ---- register-resident layer-2 weights: [W_ih2^T ; W_hh2^T], K=128 ----
  AU b2f[4][4];                         // [gate][k-frag]  (128 VGPRs)
  float b1r[4], wi1r[4], b2r[4];        // per-lane gate biases / x weight
#pragma unroll
  for (int g = 0; g < 4; ++g) {
    const int n_g = g * 64 + jcol;
#pragma unroll
    for (int kf = 0; kf < 4; ++kf)
#pragma unroll
      for (int jj = 0; jj < 16; ++jj)
        b2f[g][kf].h[jj] = (_Float16)w2_at(W_ih2, W_hh2, kf * 32 + klow + jj + (jj & 8), n_g);
    if (jcol < HID) {
      const int r = g * HID + jcol;
      b1r[g]  = b_ih1[r] + b_hh1[r];
      wi1r[g] = W_ih1[r];               // W_ih1 is [204,1]
      b2r[g]  = b_ih2[r] + b_hh2[r];
    } else { b1r[g] = 0.0f; wi1r[g] = 0.0f; b2r[g] = 0.0f; }
  }

  // register-resident cell states (C-fragment layout, rows mrow0..mrow0+7)
  float c1[8], c2[8];
#pragma unroll
  for (int r = 0; r < 8; ++r) { c1[r] = 0.0f; c2[r] = 0.0f; }

  __syncthreads();

  for (int t = 0; t < TF; ++t) {
    const int rp = t & 1;               // read (old h) buffer
    const int wp = rp ^ 1;              // write (new h) buffer

    // ---------- layer 1: acc[g] = h1_old[16,64] @ W1[64, gate g slice] ----------
    AU a0, a1;
    {
      const v8h* row = (const v8h*)&h1buf[rp][m_a * KP];
      a0.p[0] = row[abase];     a0.p[1] = row[abase + 2];   // K 0..31
      a1.p[0] = row[4 + abase]; a1.p[1] = row[6 + abase];   // K 32..63
    }
    CU acc[4];
#pragma unroll
    for (int g = 0; g < 4; ++g) {
      const v8h* wb = (const v8h*)&w1t[(g * 64 + jcol) * KP];
      AU bfr;
      v8f c = {};
      bfr.p[0] = wb[kl8];     bfr.p[1] = wb[kl8 + 2];       // K 0..31 frag
      c = wmma_f16(a0, bfr, c);
      bfr.p[0] = wb[4 + kl8]; bfr.p[1] = wb[6 + kl8];       // K 32..63 frag
      c = wmma_f16(a1, bfr, c);
      acc[g].v = c;
    }

    // ---------- layer 1 cell update, fully in registers ----------
    {
      float4 xa = *(const float4*)&xbuf[mrow0];
      float4 xb = *(const float4*)&xbuf[mrow0 + 4];
      float xr[8] = {xa.x, xa.y, xa.z, xa.w, xb.x, xb.y, xb.z, xb.w};
#pragma unroll
      for (int r = 0; r < 8; ++r) {
        float xv = xr[r];
        float pi = acc[0].f[r] + b1r[0] + xv * wi1r[0];
        float pf = acc[1].f[r] + b1r[1] + xv * wi1r[1];
        float pg = acc[2].f[r] + b1r[2] + xv * wi1r[2];
        float po = acc[3].f[r] + b1r[3] + xv * wi1r[3];
        float cn = sigf(pf) * c1[r] + sigf(pi) * tanh_fast(pg);
        float hn = sigf(po) * tanh_fast(cn);
        c1[r] = cn;
        h1buf[wp][(mrow0 + r) * KP + jcol] = (_Float16)hn;
      }
    }
    __syncthreads();   // publish h1_new across waves

    // ---------- layer 2: [h1_new | h2_old][16,128] @ W2 (register B) ----------
    AU a2, a3, a4, a5;
    {
      const v8h* r1 = (const v8h*)&h1buf[wp][m_a * KP];
      const v8h* r2 = (const v8h*)&h2buf[rp][m_a * KP];
      a2.p[0] = r1[abase];     a2.p[1] = r1[abase + 2];
      a3.p[0] = r1[4 + abase]; a3.p[1] = r1[6 + abase];
      a4.p[0] = r2[abase];     a4.p[1] = r2[abase + 2];
      a5.p[0] = r2[4 + abase]; a5.p[1] = r2[6 + abase];
    }
#pragma unroll
    for (int g = 0; g < 4; ++g) {
      v8f c = {};
      c = wmma_f16(a2, b2f[g][0], c);
      c = wmma_f16(a3, b2f[g][1], c);
      c = wmma_f16(a4, b2f[g][2], c);
      c = wmma_f16(a5, b2f[g][3], c);
      acc[g].v = c;
    }
#pragma unroll
    for (int r = 0; r < 8; ++r) {
      float pi = acc[0].f[r] + b2r[0];
      float pf = acc[1].f[r] + b2r[1];
      float pg = acc[2].f[r] + b2r[2];
      float po = acc[3].f[r] + b2r[3];
      float cn = sigf(pf) * c2[r] + sigf(pi) * tanh_fast(pg);
      float hn = sigf(po) * tanh_fast(cn);
      c2[r] = cn;
      h2buf[wp][(mrow0 + r) * KP + jcol] = (_Float16)hn;
    }
    __syncthreads();   // publish h2_new

    // ---------- projection + next-step input (wave 0, shuffle reduce) ----------
    if (js == 0) {
      const int m  = nlan;
      const int j0 = hi * 32;                      // each half-wave sums 32 of 64 K
      const v8h*    hp = (const v8h*)&h2buf[wp][m * KP + j0];
      const float4* wl = (const float4*)&wlinpad[j0];
      float s = 0.0f;
#pragma unroll
      for (int q = 0; q < 4; ++q) {
        v8h hv = hp[q];
        float4 w0 = wl[q * 2], w1v = wl[q * 2 + 1];
        s += (float)hv[0]*w0.x  + (float)hv[1]*w0.y  + (float)hv[2]*w0.z  + (float)hv[3]*w0.w
           + (float)hv[4]*w1v.x + (float)hv[5]*w1v.y + (float)hv[6]*w1v.z + (float)hv[7]*w1v.w;
      }
      s += __shfl_xor(s, 16, 32);                  // combine the two K halves
      if (lane < BT) {
        float y = s + blin;
        out[(size_t)(b0 + m) * TF + t] = y;
        const int tn = t + 1;
        if (tn < TF)
          xbuf[m] = (tn < TLEN) ? x[(size_t)(b0 + m) * TLEN + tn] : y;
      }
    }
    __syncthreads();   // xbuf ready for next step
  }
}

extern "C" void kernel_launch(void* const* d_in, const int* in_sizes, int n_in,
                              void* d_out, int out_size, void* d_ws, size_t ws_size,
                              hipStream_t stream) {
  (void)in_sizes; (void)n_in; (void)d_ws; (void)ws_size; (void)out_size;
  const float* x     = (const float*)d_in[0];
  const float* W_ih1 = (const float*)d_in[1];
  const float* W_hh1 = (const float*)d_in[2];
  const float* b_ih1 = (const float*)d_in[3];
  const float* b_hh1 = (const float*)d_in[4];
  const float* W_ih2 = (const float*)d_in[5];
  const float* W_hh2 = (const float*)d_in[6];
  const float* b_ih2 = (const float*)d_in[7];
  const float* b_hh2 = (const float*)d_in[8];
  const float* W_lin = (const float*)d_in[9];
  const float* b_lin = (const float*)d_in[10];
  const int*   fut   = (const int*)d_in[11];
  float* out = (float*)d_out;

  dim3 grid(BATCH / BT);   // 64 independent workgroups, one batch tile each
  dim3 block(NTH);         // 4 waves of 32 (wave32)
  lstm2_persistent<<<grid, block, 0, stream>>>(x, W_ih1, W_hh1, b_ih1, b_hh1,
                                               W_ih2, W_hh2, b_ih2, b_hh2,
                                               W_lin, b_lin, fut, out);
}
